// Attention_33011118637417
// MI455X (gfx1250) — matrix operand order repaired and checked by reading, not compiled, not run on hardware
//
#include <hip/hip_runtime.h>

// ---------------- problem constants ----------------
#define BB   256
#define SS   200
#define SP   224        // S padded to multiple of 32
#define DIMM 1024
#define HDD  128
#define NQT  13         // ceil(200/16) q-tiles per batch

typedef __attribute__((ext_vector_type(16))) __bf16 v16bf;
typedef __attribute__((ext_vector_type(8)))  __bf16 v8bf;
typedef __attribute__((ext_vector_type(8)))  float  v8f;
typedef __attribute__((ext_vector_type(4)))  float  v4f;

__device__ __forceinline__ __bf16 f2bf(float f) {
    unsigned u = __builtin_bit_cast(unsigned, f);
    unsigned r = u + 0x7FFFu + ((u >> 16) & 1u);   // round-to-nearest-even
    unsigned short h = (unsigned short)(r >> 16);
    return __builtin_bit_cast(__bf16, h);
}

__device__ __forceinline__ v8f wmma_bf16(v16bf a, v16bf b, v8f c) {
    return __builtin_amdgcn_wmma_f32_16x16x32_bf16(
        /*neg_a=*/false, a, /*neg_b=*/false, b,
        /*c_mod=*/(short)0, c, /*reuse_a=*/false, /*reuse_b=*/false);
}

// A-matrix (16xK bf16) fragment gather from a row-major bf16 tile.
// lanes 0-15 -> rows 0-15, K in {0..15}; lanes 16-31 same rows,
// K in {16..31}.  Two contiguous 16B LDS loads per lane.
__device__ __forceinline__ v16bf load_a_frag(const __bf16* base, int row_stride, int lane) {
    int half = lane >> 4, r = lane & 15;
    const __bf16* p = base + r * row_stride + half * 16;
    union { v16bf v; v8bf h[2]; } u;
    u.h[0] = *(const v8bf*)(p);
    u.h[1] = *(const v8bf*)(p + 8);
    return u.v;
}

// B-matrix (32x16 bf16) fragment: B[k][n] stored as rowmaj[n][k].
// Lane n = lane&15, K = 16*(lane>>4)+e -> 16 contiguous bf16 (32B load).
__device__ __forceinline__ v16bf load_b_frag(const __bf16* rowmaj, size_t row_stride,
                                             int n0, int k0, int lane) {
    const __bf16* p = rowmaj + (size_t)(n0 + (lane & 15)) * row_stride
                             + (size_t)(k0 + ((lane >> 4) << 4));
    return *(const v16bf*)p;
}

// ---- CDNA5 async global->LDS copy (16B per lane), ISA §15.18.3 op 98 ----
__device__ __forceinline__ void async_copy_b128(const void* gsrc, void* lds_dst) {
    unsigned loff = (unsigned)(unsigned long long)lds_dst;   // LDS byte offset
    asm volatile("global_load_async_to_lds_b128 %0, %1, off"
                 :: "v"(loff), "v"(gsrc) : "memory");
}
__device__ __forceinline__ void wait_async0() {
    asm volatile("s_wait_asynccnt 0" ::: "memory");
}

// ---------------- K0: weights f32 -> bf16 ----------------
__global__ __launch_bounds__(256) void k_cvt(const float* __restrict__ wq,
                                             const float* __restrict__ wk,
                                             const float* __restrict__ wv,
                                             const float* __restrict__ wo,
                                             __bf16* wqb, __bf16* wkb,
                                             __bf16* wvb, __bf16* wob) {
    int i = blockIdx.x * 256 + threadIdx.x;       // 131072 elems each
    if (i < HDD * DIMM) {
        wqb[i] = f2bf(wq[i]);
        wkb[i] = f2bf(wk[i]);
        wvb[i] = f2bf(wv[i]);
        wob[i] = f2bf(wo[i]);
    }
}

// ---------------- K1: fused QKV projection ----------------
// WG = 16 rows of flattened (B*S), 8 waves; wave w owns head-channel N-tile w
// and accumulates q,k,v together.  K staged in 128-wide chunks (4 WMMA steps
// per stage): 2 global_load_b128 + 1 ds_store_b128 per thread per stage.
__global__ __launch_bounds__(256) void k_qkv(const float* __restrict__ x,
                                             const __bf16* __restrict__ wqb,
                                             const __bf16* __restrict__ wkb,
                                             const __bf16* __restrict__ wvb,
                                             float* __restrict__ qf,
                                             float* __restrict__ kf,
                                             __bf16* __restrict__ vT) {
    __shared__ __bf16 As[16 * 128];               // 4 KB x-chunk, bf16
    const int tid  = threadIdx.x;
    const int lane = tid & 31;
    const int wave = tid >> 5;
    const int row0 = blockIdx.x * 16;             // 3200 tiles cover B*S exactly
    const int n0   = wave * 16;

    // this thread stages 8 contiguous x elements: row sm, cols [sk, sk+8)
    const int sm = tid >> 4;
    const int sk = (tid & 15) * 8;
    const float* xrow = x + (size_t)(row0 + sm) * DIMM + sk;

    v8f accq = {}, acck = {}, accv = {};

    for (int c0 = 0; c0 < DIMM; c0 += 128) {
        __syncthreads();
        {
            v4f x0 = *(const v4f*)(xrow + c0);
            v4f x1 = *(const v4f*)(xrow + c0 + 4);
            if (c0 + 128 < DIMM)
                __builtin_prefetch(xrow + c0 + 128, 0, 1);
            v8bf t;
            t[0] = f2bf(x0[0]); t[1] = f2bf(x0[1]);
            t[2] = f2bf(x0[2]); t[3] = f2bf(x0[3]);
            t[4] = f2bf(x1[0]); t[5] = f2bf(x1[1]);
            t[6] = f2bf(x1[2]); t[7] = f2bf(x1[3]);
            *(v8bf*)(&As[tid * 8]) = t;
        }
        __syncthreads();

        v16bf a0 = load_a_frag(As +  0, 128, lane);
        v16bf a1 = load_a_frag(As + 32, 128, lane);
        v16bf a2 = load_a_frag(As + 64, 128, lane);
        v16bf a3 = load_a_frag(As + 96, 128, lane);

        v16bf bq0 = load_b_frag(wqb, DIMM, n0, c0 +  0, lane);
        v16bf bq1 = load_b_frag(wqb, DIMM, n0, c0 + 32, lane);
        v16bf bq2 = load_b_frag(wqb, DIMM, n0, c0 + 64, lane);
        v16bf bq3 = load_b_frag(wqb, DIMM, n0, c0 + 96, lane);
        v16bf bk0 = load_b_frag(wkb, DIMM, n0, c0 +  0, lane);
        v16bf bk1 = load_b_frag(wkb, DIMM, n0, c0 + 32, lane);
        v16bf bk2 = load_b_frag(wkb, DIMM, n0, c0 + 64, lane);
        v16bf bk3 = load_b_frag(wkb, DIMM, n0, c0 + 96, lane);
        v16bf bv0 = load_b_frag(wvb, DIMM, n0, c0 +  0, lane);
        v16bf bv1 = load_b_frag(wvb, DIMM, n0, c0 + 32, lane);
        v16bf bv2 = load_b_frag(wvb, DIMM, n0, c0 + 64, lane);
        v16bf bv3 = load_b_frag(wvb, DIMM, n0, c0 + 96, lane);

        accq = wmma_bf16(a0, bq0, accq);
        acck = wmma_bf16(a0, bk0, acck);
        accv = wmma_bf16(a0, bv0, accv);
        accq = wmma_bf16(a1, bq1, accq);
        acck = wmma_bf16(a1, bk1, acck);
        accv = wmma_bf16(a1, bv1, accv);
        accq = wmma_bf16(a2, bq2, accq);
        acck = wmma_bf16(a2, bk2, acck);
        accv = wmma_bf16(a2, bv2, accv);
        accq = wmma_bf16(a3, bq3, accq);
        acck = wmma_bf16(a3, bk3, acck);
        accv = wmma_bf16(a3, bv3, accv);
    }

    const int half = lane >> 4, nn = lane & 15;
    #pragma unroll
    for (int r = 0; r < 8; ++r) {
        int m = r + half * 8;
        size_t g = (size_t)(row0 + m);
        int h = n0 + nn;
        qf[g * HDD + h] = accq[r];
        kf[g * HDD + h] = acck[r];
        int b = (int)(g / SS), s = (int)(g % SS);
        vT[((size_t)b * HDD + h) * SP + s] = f2bf(accv[r]);
    }
}

// ---------------- K2: RoPE + bf16 convert + zero-pad ----------------
__global__ __launch_bounds__(256) void k_rope(const float* __restrict__ qf,
                                              const float* __restrict__ kf,
                                              __bf16* __restrict__ qr,
                                              __bf16* __restrict__ kr,
                                              __bf16* __restrict__ vT) {
    size_t idx = (size_t)blockIdx.x * 256 + threadIdx.x;  // B*SP*64 threads
    int p = (int)(idx & 63);                              // channel pair
    int s = (int)((idx >> 6) % SP);
    int b = (int)(idx / ((size_t)SP * 64));
    size_t rbase = ((size_t)b * SP + s) * HDD;
    int h0 = 2 * p;

    if (s >= SS) {                                        // zero padding rows
        __bf16 z = f2bf(0.0f);
        qr[rbase + h0] = z; qr[rbase + h0 + 1] = z;
        kr[rbase + h0] = z; kr[rbase + h0 + 1] = z;
        vT[((size_t)b * HDD + h0) * SP + s]     = z;
        vT[((size_t)b * HDD + h0 + 1) * SP + s] = z;
        return;
    }
    size_t gbase = ((size_t)b * SS + s) * HDD;
    float q0 = qf[gbase + h0], q1 = qf[gbase + h0 + 1];
    float k0v = kf[gbase + h0], k1 = kf[gbase + h0 + 1];
    if (s >= 1) {
        float t   = (float)(s - 1);
        float inv = __powf(10000.0f, -(float)h0 / (float)HDD);
        float ang = t * inv;
        float c = __cosf(ang), sn = __sinf(ang);
        float r0 = q0 * c - q1 * sn, r1 = q0 * sn + q1 * c;
        q0 = r0; q1 = r1;
        r0 = k0v * c - k1 * sn; r1 = k0v * sn + k1 * c;
        k0v = r0; k1 = r1;
    }
    qr[rbase + h0]     = f2bf(q0);
    qr[rbase + h0 + 1] = f2bf(q1);
    kr[rbase + h0]     = f2bf(k0v);
    kr[rbase + h0 + 1] = f2bf(k1);
}

// ---------------- K3: attention (scores + softmax + PV) ----------------
__global__ __launch_bounds__(256) void k_attn(const __bf16* __restrict__ qr,
                                              const __bf16* __restrict__ kr,
                                              const __bf16* __restrict__ vT,
                                              __bf16* __restrict__ ao) {
    __shared__ __bf16 Qs[16 * HDD];   // 4 KB q-tile
    __shared__ float  Ss[16 * SP];    // 14 KB scores
    __shared__ __bf16 Ps[16 * SP];    // 7 KB probabilities (bf16)

    const int tid  = threadIdx.x;
    const int lane = tid & 31;
    const int wave = tid >> 5;
    const int b  = blockIdx.x / NQT;
    const int s0 = (blockIdx.x % NQT) * 16;

    // q-tile is fully contiguous in qr: async-copy 16B per thread (4 KB total)
    async_copy_b128(qr + ((size_t)b * SP + s0) * HDD + tid * 8, &Qs[tid * 8]);
    wait_async0();
    __syncthreads();

    // ---- scores = (Q Kt) / sqrt(HD), 14 N-tiles over 8 waves ----
    const __bf16* krb = kr + (size_t)b * SP * HDD;
    const float scale = 0.088388347648318447f;   // 1/sqrt(128)
    for (int t = wave; t < SP / 16; t += 8) {
        int n0 = t * 16;
        v8f acc = {};
        #pragma unroll
        for (int k0 = 0; k0 < HDD; k0 += 32) {
            v16bf a  = load_a_frag(Qs + k0, HDD, lane);
            v16bf bk = load_b_frag(krb, HDD, n0, k0, lane);
            acc = wmma_bf16(a, bk, acc);
        }
        int half = lane >> 4, nn = lane & 15;
        #pragma unroll
        for (int r = 0; r < 8; ++r) {
            int m = r + half * 8, col = n0 + nn;
            Ss[m * SP + col] = (col < SS) ? acc[r] * scale : -1.0e30f;
        }
    }
    __syncthreads();

    // ---- softmax: 16 threads per row, half-wave shuffle reductions ----
    {
        int row = tid >> 4, c = tid & 15;
        float mx = -1.0e30f;
        for (int j = c; j < SP; j += 16) mx = fmaxf(mx, Ss[row * SP + j]);
        #pragma unroll
        for (int o = 8; o >= 1; o >>= 1) mx = fmaxf(mx, __shfl_xor(mx, o, 32));
        float sum = 0.0f;
        for (int j = c; j < SP; j += 16) {
            float e = __expf(Ss[row * SP + j] - mx);
            Ss[row * SP + j] = e;
            sum += e;
        }
        #pragma unroll
        for (int o = 8; o >= 1; o >>= 1) sum += __shfl_xor(sum, o, 32);
        float inv = 1.0f / sum;
        for (int j = c; j < SP; j += 16) Ps[row * SP + j] = f2bf(Ss[row * SP + j] * inv);
    }
    __syncthreads();

    // ---- out = P @ V : wave w owns N-tile w of the 128 head channels ----
    {
        const __bf16* vtb = vT + (size_t)b * HDD * SP;
        int n0 = wave * 16;
        v8f acc = {};
        #pragma unroll
        for (int k0 = 0; k0 < SP; k0 += 32) {
            v16bf a  = load_a_frag(Ps + k0, SP, lane);
            v16bf bv = load_b_frag(vtb, SP, n0, k0, lane);
            acc = wmma_bf16(a, bv, acc);
        }
        int half = lane >> 4, nn = lane & 15;
        #pragma unroll
        for (int r = 0; r < 8; ++r) {
            int m = r + half * 8;
            ao[((size_t)b * SP + s0 + m) * HDD + n0 + nn] = f2bf(acc[r]);
        }
    }
}

// ---------------- K4: output projection (attn @ Wo^T) ----------------
__global__ __launch_bounds__(256) void k_out(const __bf16* __restrict__ ao,
                                             const __bf16* __restrict__ wob,
                                             float* __restrict__ out) {
    __shared__ __bf16 As[16 * HDD];   // 4 KB attn-out tile
    const int tid  = threadIdx.x;
    const int lane = tid & 31;
    const int wave = tid >> 5;
    const int b  = blockIdx.x / NQT;
    const int s0 = (blockIdx.x % NQT) * 16;

    async_copy_b128(ao + ((size_t)b * SP + s0) * HDD + tid * 8, &As[tid * 8]);
    wait_async0();
    __syncthreads();

    for (int t = wave; t < DIMM / 16; t += 8) {     // 64 N-tiles over 8 waves
        int n0 = t * 16;
        v8f acc = {};
        #pragma unroll
        for (int k0 = 0; k0 < HDD; k0 += 32) {
            v16bf a  = load_a_frag(As + k0, HDD, lane);
            v16bf bw = load_b_frag(wob, HDD, n0, k0, lane);
            acc = wmma_bf16(a, bw, acc);
        }
        int half = lane >> 4, nn = lane & 15;
        #pragma unroll
        for (int r = 0; r < 8; ++r) {
            int m = r + half * 8, s = s0 + m;
            if (s < SS)
                out[((size_t)b * SS + s) * DIMM + n0 + nn] = acc[r];
        }
    }
}

// ---------------- launch ----------------
extern "C" void kernel_launch(void* const* d_in, const int* in_sizes, int n_in,
                              void* d_out, int out_size, void* d_ws, size_t ws_size,
                              hipStream_t stream) {
    const float* x  = (const float*)d_in[0];
    const float* wq = (const float*)d_in[1];
    const float* wk = (const float*)d_in[2];
    const float* wv = (const float*)d_in[3];
    const float* wo = (const float*)d_in[4];
    float* out = (float*)d_out;

    char* ws = (char*)d_ws;
    size_t o = 0;
    const size_t WB = (size_t)HDD * DIMM * sizeof(__bf16);      // 256 KB
    __bf16* wqb = (__bf16*)(ws + o); o += WB;
    __bf16* wkb = (__bf16*)(ws + o); o += WB;
    __bf16* wvb = (__bf16*)(ws + o); o += WB;
    __bf16* wob = (__bf16*)(ws + o); o += WB;
    float*  qf  = (float*)(ws + o);  o += (size_t)BB * SS * HDD * sizeof(float);
    float*  kf  = (float*)(ws + o);  o += (size_t)BB * SS * HDD * sizeof(float);
    const size_t PB = (size_t)BB * SP * HDD * sizeof(__bf16);   // 14 MB each
    __bf16* qr  = (__bf16*)(ws + o); o += PB;
    __bf16* kr  = (__bf16*)(ws + o); o += PB;
    __bf16* vT  = (__bf16*)(ws + o); o += PB;
    __bf16* ao  = (__bf16*)(ws + o); o += PB;

    k_cvt <<<(HDD * DIMM + 255) / 256, 256, 0, stream>>>(wq, wk, wv, wo,
                                                         wqb, wkb, wvb, wob);
    k_qkv <<<(BB * SS) / 16, 256, 0, stream>>>(x, wqb, wkb, wvb, qf, kf, vT);
    k_rope<<<(BB * SP * (HDD / 2)) / 256, 256, 0, stream>>>(qf, kf, qr, kr, vT);
    k_attn<<<BB * NQT, 256, 0, stream>>>(qr, kr, vT, ao);
    k_out <<<BB * NQT, 256, 0, stream>>>(ao, wob, out);
}
